// MMoE_64098091925661
// MI455X (gfx1250) — compile-verified, matrix operand order checked by reading, not run on hardware
//
#include <hip/hip_runtime.h>
#include <hip/hip_bf16.h>

typedef __bf16 bf16;
typedef bf16  bf16x8  __attribute__((ext_vector_type(8)));
typedef bf16  bf16x16 __attribute__((ext_vector_type(16)));
typedef float floatx8 __attribute__((ext_vector_type(8)));

#define NB_  8192   // batch
#define NF_  1024   // features
#define NE_  20     // experts
#define NH1_ 512
#define NH2_ 256
#define NT_  10     // tasks
#define NT1_ 128
#define NT2_ 64
#define NGP_ 256    // gate logits padded (T*E=200 -> 256, multiple of 64)

#define BPITCH 80   // LDS row pitch for a 64B row of B tile (16B aligned, bank-friendly)

// ---------------------------------------------------------------------------
// BF16 WMMA GEMM:  C[M,N] = A[M,K] (bf16 row-major) x Bt[N,K]^T
// Block: 4 waves stacked in M -> block tile 256(M) x 64(N).
// Wave tile 64x64: 4x4 accumulators, 16 v_wmma_f32_16x16x32_bf16 per K=32.
// B tile (64 rows x 32 k) is staged cooperatively into LDS with
// global_load_async_to_lds_b128 (double buffered, s_wait_asynccnt paced);
// A fragments are global->reg, software-prefetched one K-step ahead.
// mode 0: +bias -> f32.  mode 1: +bias, eval-BN + ReLU -> bf16.
// ---------------------------------------------------------------------------
struct AFrag { bf16x16 f[4]; };

__device__ __forceinline__ AFrag load_a(const bf16* const arow[4], int k0)
{
    AFrag r;
#pragma unroll
    for (int i = 0; i < 4; ++i) {
        bf16x8 lo = *(const bf16x8*)(arow[i] + k0);
        bf16x8 hi = *(const bf16x8*)(arow[i] + k0 + 16);
        r.f[i] = __builtin_shufflevector(lo, hi, 0,1,2,3,4,5,6,7,8,9,10,11,12,13,14,15);
    }
    return r;
}

// Copy 64 rows x 64 bytes (one K=32 chunk of the B tile) into LDS.
// 256 16B chunks, 128 threads -> 2 async b128 copies per thread (+2 ASYNCcnt).
__device__ __forceinline__ void async_copy_B(const bf16* __restrict__ Bb, int K,
                                             int kk, char* lp, int tid)
{
#pragma unroll
    for (int u = 0; u < 2; ++u) {
        const int ch   = tid + u * 128;
        const int row  = ch >> 2;
        const int part = ch & 3;
        const unsigned long long g =
            (unsigned long long)(size_t)(Bb + (size_t)row * K + kk + part * 8);
        const unsigned lds = (unsigned)(size_t)(lp + row * BPITCH + part * 16);
        asm volatile("global_load_async_to_lds_b128 %0, %1, off"
                     : : "v"(lds), "v"(g) : "memory");
    }
}

__device__ __forceinline__ void mma_step(const char* bufrow, int l15, int hl,
                                         const AFrag& a, floatx8 acc[4][4])
{
    bf16x16 bfr[4];
#pragma unroll
    for (int j = 0; j < 4; ++j) {
        const char* rb = bufrow + (size_t)(j * 16 + l15) * BPITCH + hl * 16;
        bf16x8 lo = *(const bf16x8*)rb;
        bf16x8 hi = *(const bf16x8*)(rb + 32);
        bfr[j] = __builtin_shufflevector(lo, hi, 0,1,2,3,4,5,6,7,8,9,10,11,12,13,14,15);
    }
#pragma unroll
    for (int i = 0; i < 4; ++i)
#pragma unroll
        for (int j = 0; j < 4; ++j)
            acc[i][j] = __builtin_amdgcn_wmma_f32_16x16x32_bf16(
                false, a.f[i], false, bfr[j], (short)0, acc[i][j], false, false);
}

__global__ __launch_bounds__(128)
void gemm_bf16(const bf16* __restrict__ A, long long sA,
               const bf16* __restrict__ Bt, long long sB,
               const float* __restrict__ bias, int sBias,
               const float* __restrict__ bng, const float* __restrict__ bnb,
               const float* __restrict__ bnm, const float* __restrict__ bnv, int sBn,
               void* __restrict__ C, long long sC,
               int M, int N, int K, int mode)
{
    __shared__ __align__(16) char bufB[2][64 * BPITCH];

    const int bt   = blockIdx.z;
    const int tid  = threadIdx.x;
    const int lane = tid & 31;
    const int wave = tid >> 5;
    const int l15  = lane & 15;
    const int hl   = lane >> 4;
    const int tileN = blockIdx.x * 64;
    const int tileM = blockIdx.y * 256 + wave * 64;

    const bf16* Ab = A  + (size_t)bt * (size_t)sA;
    const bf16* Bb = Bt + (size_t)bt * (size_t)sB + (size_t)tileN * K;

    const bf16* arow[4];
#pragma unroll
    for (int i = 0; i < 4; ++i) arow[i] = Ab + (size_t)(tileM + i * 16 + l15) * K;

    floatx8 acc[4][4];
#pragma unroll
    for (int i = 0; i < 4; ++i)
#pragma unroll
        for (int j = 0; j < 4; ++j)
#pragma unroll
            for (int r = 0; r < 8; ++r) acc[i][j][r] = 0.0f;

    const int kh = hl * 8;
    AFrag afA = load_a(arow, kh);
    async_copy_B(Bb, K, 0, &bufB[0][0], tid);

    for (int k = 0; k < K; k += 64) {            // K is always a multiple of 64
        // ---- step 1: prefetch (k+32), compute (k) out of buf0 ----
        async_copy_B(Bb, K, k + 32, &bufB[1][0], tid);
        AFrag afB = load_a(arow, k + 32 + kh);
        asm volatile("s_wait_asynccnt 0x2" ::: "memory");   // buf0 copies done
        __syncthreads();
        mma_step(&bufB[0][0], l15, hl, afA, acc);
        __syncthreads();

        // ---- step 2: prefetch (k+64), compute (k+32) out of buf1 ----
        const bool more = (k + 64) < K;
        if (more) {
            async_copy_B(Bb, K, k + 64, &bufB[0][0], tid);
            afA = load_a(arow, k + 64 + kh);
            asm volatile("s_wait_asynccnt 0x2" ::: "memory");
        } else {
            asm volatile("s_wait_asynccnt 0x0" ::: "memory");
        }
        __syncthreads();
        mma_step(&bufB[1][0], l15, hl, afB, acc);
        __syncthreads();
    }

    // Epilogue.  C/D layout: VGPR r -> row = r + 8*(lane>=16), col = lane&15.
#pragma unroll
    for (int j = 0; j < 4; ++j) {
        const int n = tileN + j * 16 + l15;
        const float bia = bias ? bias[(size_t)bt * sBias + n] : 0.0f;
        float sc = 1.0f, sh = 0.0f;
        if (mode == 1) {
            const float g  = bng[(size_t)bt * sBn + n];
            const float be = bnb[(size_t)bt * sBn + n];
            const float mm = bnm[(size_t)bt * sBn + n];
            const float vv = bnv[(size_t)bt * sBn + n];
            sc = g * rsqrtf(vv + 1e-5f);
            sh = be - mm * sc;
        }
#pragma unroll
        for (int i = 0; i < 4; ++i) {
#pragma unroll
            for (int r = 0; r < 8; ++r) {
                const int m = tileM + i * 16 + hl * 8 + r;
                float v = acc[i][j][r] + bia;
                if (mode == 1) {
                    v = fmaxf(v * sc + sh, 0.0f);
                    ((bf16*)C)[(size_t)bt * sC + (size_t)m * N + n] = (bf16)v;
                } else {
                    ((float*)C)[(size_t)bt * sC + (size_t)m * N + n] = v;
                }
            }
        }
    }
}

// --------------------------- prep kernels ----------------------------------
__global__ void convert_f32_bf16(const float* __restrict__ in, bf16* __restrict__ out, size_t n)
{
    size_t i = ((size_t)blockIdx.x * blockDim.x + threadIdx.x) * 4;
    const size_t stride = (size_t)gridDim.x * blockDim.x * 4;
    for (; i + 3 < n; i += stride) {
        float4 v = *(const float4*)(in + i);
        out[i + 0] = (bf16)v.x; out[i + 1] = (bf16)v.y;
        out[i + 2] = (bf16)v.z; out[i + 3] = (bf16)v.w;
    }
}

// in: [bt][K][N] f32  ->  out: [bt][N][K] bf16  (LDS-tiled so both sides coalesce)
__global__ void transpose_convert(const float* __restrict__ in, bf16* __restrict__ out,
                                  int K, int N)
{
    __shared__ float tile[32][33];
    const int bt = blockIdx.z;
    in  += (size_t)bt * K * N;
    out += (size_t)bt * N * K;
    const int k0 = blockIdx.y * 32, n0 = blockIdx.x * 32;
    const int tx = threadIdx.x, ty = threadIdx.y;   // (32, 8)
#pragma unroll
    for (int i = 0; i < 32; i += 8) {
        const int k = k0 + ty + i, n = n0 + tx;
        tile[ty + i][tx] = (k < K && n < N) ? in[(size_t)k * N + n] : 0.0f;
    }
    __syncthreads();
#pragma unroll
    for (int i = 0; i < 32; i += 8) {
        const int n = n0 + ty + i, k = k0 + tx;
        if (n < N && k < K) out[(size_t)n * K + k] = (bf16)tile[tx][ty + i];
    }
}

// gW [T][F][E] f32 -> gwt [256][F] bf16 (n = t*20+e, rows >=200 zeroed)
// gb [T][E]        -> gbias[256] f32
__global__ void pack_gates(const float* __restrict__ gW, const float* __restrict__ gb,
                           bf16* __restrict__ gwt, float* __restrict__ gbias)
{
    const int n = blockIdx.y;
    const int k = blockIdx.x * blockDim.x + threadIdx.x;
    const int t = n / NE_, e = n % NE_;
    const float v = (n < NT_ * NE_) ? gW[((size_t)t * NF_ + k) * NE_ + e] : 0.0f;
    gwt[(size_t)n * NF_ + k] = (bf16)v;
    if (k == 0) gbias[n] = (n < NT_ * NE_) ? gb[t * NE_ + e] : 0.0f;
}

// ------------------------- small VALU kernels ------------------------------
__global__ void gate_softmax(const float* __restrict__ logits, float* __restrict__ gates)
{
    const int idx = blockIdx.x * blockDim.x + threadIdx.x;   // t*B + b
    if (idx >= NT_ * NB_) return;
    const int t = idx / NB_, b = idx % NB_;
    const float* p = logits + (size_t)b * NGP_ + t * NE_;
    float mx = -3.4e38f;
#pragma unroll
    for (int e = 0; e < NE_; ++e) mx = fmaxf(mx, p[e]);
    float ex[NE_], s = 0.0f;
#pragma unroll
    for (int e = 0; e < NE_; ++e) { ex[e] = __expf(p[e] - mx); s += ex[e]; }
    const float inv = 1.0f / s;
    float* g = gates + (size_t)idx * NE_;
#pragma unroll
    for (int e = 0; e < NE_; ++e) g[e] = ex[e] * inv;
}

// task_fea[t,b,:] = sum_e gate[t,b,e] * fea[e,b,:]   (one thread per 8 dims)
__global__ void combine(const float* __restrict__ gates, const bf16* __restrict__ fea,
                        bf16* __restrict__ tfea)
{
    const int idx = blockIdx.x * blockDim.x + threadIdx.x;
    const int d8 = idx & (NH2_ / 8 - 1);
    const int tb = idx / (NH2_ / 8);
    if (tb >= NT_ * NB_) return;
    const int b = tb % NB_;
    const float* g = gates + (size_t)tb * NE_;
    float accv[8];
#pragma unroll
    for (int j = 0; j < 8; ++j) accv[j] = 0.0f;
    for (int e = 0; e < NE_; ++e) {
        const bf16x8 f = *(const bf16x8*)(fea + ((size_t)e * NB_ + b) * NH2_ + d8 * 8);
        const float ge = g[e];
#pragma unroll
        for (int j = 0; j < 8; ++j) accv[j] += ge * (float)f[j];
    }
    bf16x8 o;
#pragma unroll
    for (int j = 0; j < 8; ++j) o[j] = (bf16)accv[j];
    *(bf16x8*)(tfea + (size_t)tb * NH2_ + d8 * 8) = o;
}

// Final per-task heads: out_t = t2[t] @ tW3[t,:, :c_t] + tb3[t, :c_t]
__global__ void tower3(const bf16* __restrict__ t2, const float* __restrict__ tW3,
                       const float* __restrict__ tb3, float* __restrict__ out)
{
    const int idx = blockIdx.x * blockDim.x + threadIdx.x;   // t*B + b
    if (idx >= NT_ * NB_) return;
    const int t = idx / NB_, b = idx % NB_;
    const int cs[NT_]   = {6, 5, 5, 2, 2, 2, 2, 2, 2, 2};
    const int offs[NT_] = {0, 6, 11, 16, 18, 20, 22, 24, 26, 28};
    const int c = cs[t];
    const bf16* a = t2 + (size_t)idx * NT2_;
    float av[NT2_];
#pragma unroll
    for (int k = 0; k < NT2_; ++k) av[k] = (float)a[k];
    for (int n = 0; n < c; ++n) {
        float s = tb3[t * 6 + n];
#pragma unroll
        for (int k = 0; k < NT2_; ++k) s += av[k] * tW3[((size_t)t * NT2_ + k) * 6 + n];
        out[(size_t)offs[t] * NB_ + (size_t)b * c + n] = s;
    }
}

// ---------------------------------------------------------------------------
extern "C" void kernel_launch(void* const* d_in, const int* in_sizes, int n_in,
                              void* d_out, int out_size, void* d_ws, size_t ws_size,
                              hipStream_t stream)
{
    const float* x      = (const float*)d_in[0];
    const float* eW1    = (const float*)d_in[1];
    const float* eb1    = (const float*)d_in[2];
    const float* ebn1_g = (const float*)d_in[3];
    const float* ebn1_b = (const float*)d_in[4];
    const float* ebn1_m = (const float*)d_in[5];
    const float* ebn1_v = (const float*)d_in[6];
    const float* eW2    = (const float*)d_in[7];
    const float* eb2    = (const float*)d_in[8];
    const float* ebn2_g = (const float*)d_in[9];
    const float* ebn2_b = (const float*)d_in[10];
    const float* ebn2_m = (const float*)d_in[11];
    const float* ebn2_v = (const float*)d_in[12];
    const float* gW     = (const float*)d_in[13];
    const float* gb     = (const float*)d_in[14];
    const float* tW1    = (const float*)d_in[15];
    const float* tb1    = (const float*)d_in[16];
    const float* tbn1_g = (const float*)d_in[17];
    const float* tbn1_b = (const float*)d_in[18];
    const float* tbn1_m = (const float*)d_in[19];
    const float* tbn1_v = (const float*)d_in[20];
    const float* tW2    = (const float*)d_in[21];
    const float* tb2    = (const float*)d_in[22];
    const float* tbn2_g = (const float*)d_in[23];
    const float* tbn2_b = (const float*)d_in[24];
    const float* tbn2_m = (const float*)d_in[25];
    const float* tbn2_v = (const float*)d_in[26];
    const float* tW3    = (const float*)d_in[27];
    const float* tb3    = (const float*)d_in[28];
    float* out = (float*)d_out;

    // ---- workspace carve-up (256B aligned) ----
    char* ws = (char*)d_ws;
    size_t off = 0;
    auto alloc = [&](size_t bytes) -> void* {
        void* p = ws + off;
        off += (bytes + 255) & ~(size_t)255;
        return p;
    };
    bf16*  xb      = (bf16*) alloc((size_t)NB_ * NF_ * 2);
    bf16*  w1t     = (bf16*) alloc((size_t)NE_ * NH1_ * NF_ * 2);
    bf16*  w2t     = (bf16*) alloc((size_t)NE_ * NH2_ * NH1_ * 2);
    bf16*  gwt     = (bf16*) alloc((size_t)NGP_ * NF_ * 2);
    float* gbias   = (float*)alloc((size_t)NGP_ * 4);
    bf16*  twr1t   = (bf16*) alloc((size_t)NT_ * NT1_ * NH2_ * 2);
    bf16*  twr2t   = (bf16*) alloc((size_t)NT_ * NT2_ * NT1_ * 2);
    float* glogits = (float*)alloc((size_t)NB_ * NGP_ * 4);
    float* gates   = (float*)alloc((size_t)NT_ * NB_ * NE_ * 4);
    bf16*  h1      = (bf16*) alloc((size_t)NB_ * NH1_ * 2);           // reused per expert
    bf16*  fea     = (bf16*) alloc((size_t)NE_ * NB_ * NH2_ * 2);
    bf16*  tfea    = (bf16*) alloc((size_t)NT_ * NB_ * NH2_ * 2);
    bf16*  t1b     = (bf16*) alloc((size_t)NT_ * NB_ * NT1_ * 2);
    bf16*  t2b     = (bf16*) alloc((size_t)NT_ * NB_ * NT2_ * 2);
    (void)ws_size; (void)in_sizes; (void)n_in; (void)out_size;

    const dim3 tb32(32, 8);

    // ---- prep: convert / transpose to bf16 [N][K] ----
    convert_f32_bf16<<<1024, 256, 0, stream>>>(x, xb, (size_t)NB_ * NF_);
    transpose_convert<<<dim3(NH1_/32, NF_/32,  NE_), tb32, 0, stream>>>(eW1, w1t, NF_,  NH1_);
    transpose_convert<<<dim3(NH2_/32, NH1_/32, NE_), tb32, 0, stream>>>(eW2, w2t, NH1_, NH2_);
    transpose_convert<<<dim3(NT1_/32, NH2_/32, NT_), tb32, 0, stream>>>(tW1, twr1t, NH2_, NT1_);
    transpose_convert<<<dim3(NT2_/32, NT1_/32, NT_), tb32, 0, stream>>>(tW2, twr2t, NT1_, NT2_);
    pack_gates<<<dim3(NF_/256, NGP_), 256, 0, stream>>>(gW, gb, gwt, gbias);

    // ---- gates: [B,F] x [F,256] -> logits, then softmax over 20 experts ----
    gemm_bf16<<<dim3(NGP_/64, NB_/256, 1), 128, 0, stream>>>(
        xb, 0, gwt, 0, gbias, 0, nullptr, nullptr, nullptr, nullptr, 0,
        glogits, 0, NB_, NGP_, NF_, 0);
    gate_softmax<<<(NT_*NB_ + 255)/256, 256, 0, stream>>>(glogits, gates);

    // ---- experts: L1 (F->H1, BN+ReLU) then L2 (H1->H2, BN+ReLU) ----
    for (int e = 0; e < NE_; ++e) {
        gemm_bf16<<<dim3(NH1_/64, NB_/256, 1), 128, 0, stream>>>(
            xb, 0, w1t + (size_t)e * NH1_ * NF_, 0,
            eb1 + (size_t)e * NH1_, 0,
            ebn1_g + (size_t)e * NH1_, ebn1_b + (size_t)e * NH1_,
            ebn1_m + (size_t)e * NH1_, ebn1_v + (size_t)e * NH1_, 0,
            h1, 0, NB_, NH1_, NF_, 1);
        gemm_bf16<<<dim3(NH2_/64, NB_/256, 1), 128, 0, stream>>>(
            h1, 0, w2t + (size_t)e * NH2_ * NH1_, 0,
            eb2 + (size_t)e * NH2_, 0,
            ebn2_g + (size_t)e * NH2_, ebn2_b + (size_t)e * NH2_,
            ebn2_m + (size_t)e * NH2_, ebn2_v + (size_t)e * NH2_, 0,
            fea + (size_t)e * NB_ * NH2_, 0, NB_, NH2_, NH1_, 1);
    }

    // ---- gate-weighted combine -> task_fea [T,B,H2] ----
    combine<<<((NT_*NB_*(NH2_/8)) + 255)/256, 256, 0, stream>>>(gates, fea, tfea);

    // ---- towers, batched over T via grid.z ----
    gemm_bf16<<<dim3(NT1_/64, NB_/256, NT_), 128, 0, stream>>>(
        tfea, (long long)NB_ * NH2_, twr1t, (long long)NT1_ * NH2_,
        tb1, NT1_, tbn1_g, tbn1_b, tbn1_m, tbn1_v, NT1_,
        t1b, (long long)NB_ * NT1_, NB_, NT1_, NH2_, 1);
    gemm_bf16<<<dim3(NT2_/64, NB_/256, NT_), 128, 0, stream>>>(
        t1b, (long long)NB_ * NT1_, twr2t, (long long)NT2_ * NT1_,
        tb2, NT2_, tbn2_g, tbn2_b, tbn2_m, tbn2_v, NT2_,
        t2b, (long long)NB_ * NT2_, NB_, NT2_, NT1_, 1);

    // ---- per-task heads -> concatenated output ----
    tower3<<<(NT_*NB_ + 127)/128, 128, 0, stream>>>(t2b, tW3, tb3, out);
}
